// SOLOv2Head_10084583211138
// MI455X (gfx1250) — 8-bit matrix operand order repaired and checked by reading, compiled, not run on hardware
//
#include <hip/hip_runtime.h>
#include <hip/hip_bf16.h>

#define NMASK   200
#define KDIM    200704
#define NPAD    224
#define NBT     7
#define KSLICES 49
#define KSLICE  (KDIM / KSLICES)
#define KSTEPS  (KSLICE / 128)

#define A8_BYTES   ((size_t)NPAD * KDIM)
#define PART_ELEMS ((size_t)KSLICES * NPAD * NPAD)
#define PART_BYTES (PART_ELEMS * sizeof(float))
#define INTER_OFF  (A8_BYTES + PART_BYTES)

typedef int   v16i __attribute__((ext_vector_type(16)));
typedef float v8f  __attribute__((ext_vector_type(8)));
typedef float v4f  __attribute__((ext_vector_type(4)));

__global__ void convert_fp8_kernel(const float* __restrict__ src,
                                   unsigned char* __restrict__ dst) {
    size_t t = (size_t)blockIdx.x * blockDim.x + threadIdx.x;
    size_t byteIdx = t * 16;
    size_t row = byteIdx / KDIM;
    size_t k   = byteIdx % KDIM;
    uint4 outv;
    if (row < NMASK) {
        const v4f* p = (const v4f*)(src + row * (size_t)KDIM + k);
        unsigned int w[4];
#pragma unroll
        for (int q = 0; q < 4; ++q) {
            v4f f = __builtin_nontemporal_load(p + q);
            unsigned int b0 = (f.x > 0.5f) ? 0x38u : 0u;
            unsigned int b1 = (f.y > 0.5f) ? 0x38u : 0u;
            unsigned int b2 = (f.z > 0.5f) ? 0x38u : 0u;
            unsigned int b3 = (f.w > 0.5f) ? 0x38u : 0u;
            w[q] = b0 | (b1 << 8) | (b2 << 16) | (b3 << 24);
        }
        outv = make_uint4(w[0], w[1], w[2], w[3]);
    } else {
        outv = make_uint4(0u, 0u, 0u, 0u);
    }
    ((uint4*)dst)[t] = outv;
}

static __device__ __forceinline__ v16i load_a_frag(const unsigned char* ap) {
    union { v16i v; uint4 u4[4]; } ua;
#pragma unroll
    for (int c = 0; c < 4; ++c) {
        ua.u4[c] = *(const uint4*)(ap + 32 * c);
    }
    return ua.v;
}
static __device__ __forceinline__ v16i load_b_frag(const unsigned char* bp) {
    union { v16i v; uint4 u4[4]; } ub;
#pragma unroll
    for (int q = 0; q < 4; ++q) {
        ub.u4[q] = *(const uint4*)(bp + 32 * q);
    }
    return ub.v;
}

#define WMMA_FP8(a, b, c) \
    __builtin_amdgcn_wmma_f32_16x16x128_fp8_fp8((a), (b), (short)0, (c), false, false)

__global__ __launch_bounds__(32) void gram_fp8_kernel(
        const unsigned char* __restrict__ A,
        float* __restrict__ partials) {
    const int tile = blockIdx.x;
    const int ti = tile / NBT;
    const int tj = tile % NBT;
    const int ks = blockIdx.y;

    const int lane = threadIdx.x;
    const int g = lane >> 4;
    const int m = lane & 15;

    const size_t k0 = (size_t)ks * KSLICE;
    const unsigned char* pa0 = A + (size_t)(ti * 32 + m) * KDIM + k0 + 16 * g;
    const unsigned char* pa1 = pa0 + (size_t)16 * KDIM;
    const unsigned char* pb0 = A + (size_t)(tj * 32 + m) * KDIM + k0 + 16 * g;
    const unsigned char* pb1 = pb0 + (size_t)16 * KDIM;

    v8f acc00 = {}, acc01 = {}, acc10 = {}, acc11 = {};

    v16i a0c = load_a_frag(pa0);
    v16i a1c = load_a_frag(pa1);
    v16i b0c = load_b_frag(pb0);
    v16i b1c = load_b_frag(pb1);

#pragma unroll 1
    for (int it = 0; it < KSTEPS / 2 - 1; ++it) {
        const int kb = it * 256;
        v16i a0n = load_a_frag(pa0 + kb + 128);
        v16i a1n = load_a_frag(pa1 + kb + 128);
        v16i b0n = load_b_frag(pb0 + kb + 128);
        v16i b1n = load_b_frag(pb1 + kb + 128);
        acc00 = WMMA_FP8(a0c, b0c, acc00);
        acc01 = WMMA_FP8(a0c, b1c, acc01);
        acc10 = WMMA_FP8(a1c, b0c, acc10);
        acc11 = WMMA_FP8(a1c, b1c, acc11);
        a0c = load_a_frag(pa0 + kb + 256);
        a1c = load_a_frag(pa1 + kb + 256);
        b0c = load_b_frag(pb0 + kb + 256);
        b1c = load_b_frag(pb1 + kb + 256);
        acc00 = WMMA_FP8(a0n, b0n, acc00);
        acc01 = WMMA_FP8(a0n, b1n, acc01);
        acc10 = WMMA_FP8(a1n, b0n, acc10);
        acc11 = WMMA_FP8(a1n, b1n, acc11);
    }
    {
        const int kb = (KSTEPS - 1) * 128;
        v16i a0n = load_a_frag(pa0 + kb);
        v16i a1n = load_a_frag(pa1 + kb);
        v16i b0n = load_b_frag(pb0 + kb);
        v16i b1n = load_b_frag(pb1 + kb);
        acc00 = WMMA_FP8(a0c, b0c, acc00);
        acc01 = WMMA_FP8(a0c, b1c, acc01);
        acc10 = WMMA_FP8(a1c, b0c, acc10);
        acc11 = WMMA_FP8(a1c, b1c, acc11);
        acc00 = WMMA_FP8(a0n, b0n, acc00);
        acc01 = WMMA_FP8(a0n, b1n, acc01);
        acc10 = WMMA_FP8(a1n, b0n, acc10);
        acc11 = WMMA_FP8(a1n, b1n, acc11);
    }

    float* dst = partials + (size_t)ks * (NPAD * NPAD);
    const int r0 = ti * 32 + 8 * g;
    const int c0 = tj * 32 + m;
#pragma unroll
    for (int r = 0; r < 8; ++r) {
        dst[(size_t)(r0 + r) * NPAD + (c0)]           = acc00[r];
        dst[(size_t)(r0 + r) * NPAD + (c0 + 16)]      = acc01[r];
        dst[(size_t)(r0 + 16 + r) * NPAD + (c0)]      = acc10[r];
        dst[(size_t)(r0 + 16 + r) * NPAD + (c0 + 16)] = acc11[r];
    }
}

__global__ void reduce_partials_kernel(const float* __restrict__ partials,
                                       float* __restrict__ inter) {
    int idx = blockIdx.x * blockDim.x + threadIdx.x;
    float s = 0.f;
    for (int ks = 0; ks < KSLICES; ++ks) {
        s += partials[(size_t)ks * (NPAD * NPAD) + idx];
    }
    inter[idx] = s;
}

__global__ void matrix_nms_kernel(const float* __restrict__ inter,
                                  const float* __restrict__ scores,
                                  const int*   __restrict__ labels,
                                  float*       __restrict__ out) {
    __shared__ float s_sum[NMASK];
    __shared__ float s_comp[NMASK];
    __shared__ int   s_lab[NMASK];

    int j = threadIdx.x;
    if (j < NMASK) {
        s_lab[j] = labels[j];
        s_sum[j] = inter[(size_t)j * NPAD + j];
    }
    __syncthreads();

    if (j < NMASK) {
        const int   lj = s_lab[j];
        const float sj = s_sum[j];
        float comp = 0.f;
        for (int i = 0; i < j; ++i) {
            if (s_lab[i] == lj) {
                float in  = inter[(size_t)i * NPAD + j];
                float iou = in / (s_sum[i] + sj - in);
                comp = fmaxf(comp, iou);
            }
        }
        s_comp[j] = comp;
    }
    __syncthreads();

    if (j < NMASK) {
        const int   lj = s_lab[j];
        const float sj = s_sum[j];
        float mmin = 3.402823466e38f;
        for (int i = 0; i < NMASK; ++i) {
            float d = 0.f;
            if (i < j && s_lab[i] == lj) {
                float in = inter[(size_t)i * NPAD + j];
                d = in / (s_sum[i] + sj - in);
            }
            float v = s_comp[i] * s_comp[i] - d * d;
            mmin = fminf(mmin, v);
        }
        out[j] = scores[j] * expf(2.0f * mmin);
    }
}

extern "C" void kernel_launch(void* const* d_in, const int* in_sizes, int n_in,
                              void* d_out, int out_size, void* d_ws, size_t ws_size,
                              hipStream_t stream) {
    const float* seg    = (const float*)d_in[0];
    const float* scores = (const float*)d_in[1];
    const int*   labels = (const int*)d_in[2];
    float*       out    = (float*)d_out;

    unsigned char* ws       = (unsigned char*)d_ws;
    unsigned char* a8       = ws;
    float*         partials = (float*)(ws + A8_BYTES);
    float*         inter    = (float*)(ws + INTER_OFF);

    {
        int threads = 256;
        int blocks  = (int)(A8_BYTES / 16 / threads);
        convert_fp8_kernel<<<blocks, threads, 0, stream>>>(seg, a8);
    }
    {
        dim3 grid(NBT * NBT, KSLICES);
        gram_fp8_kernel<<<grid, 32, 0, stream>>>(a8, partials);
    }
    {
        reduce_partials_kernel<<<(NPAD * NPAD) / 256, 256, 0, stream>>>(partials, inter);
    }
    {
        matrix_nms_kernel<<<1, 256, 0, stream>>>(inter, scores, labels, out);
    }
}
